// bilinear_layer_69037304316479
// MI455X (gfx1250) — matrix operand order repaired and checked by reading, not compiled, not run on hardware
//
#include <hip/hip_runtime.h>
#include <hip/hip_bf16.h>
#include <math.h>

typedef __attribute__((ext_vector_type(16))) __bf16 v16bf;
typedef __attribute__((ext_vector_type(8)))  __bf16 v8bf;
typedef __attribute__((ext_vector_type(8)))  float  v8f;

#define B_   64
#define C_   768
#define D2_  196
#define PPAD 208      // 13 p-tiles of 16 (rows 196..207 zero-padded)
#define N_   64
#define R_   32

// ---- async-staging pipeline geometry ----
#define STAGE_CH 8                  // k-chunks (of 32 c) per stage
#define STAGE_C  (STAGE_CH * 32)    // 256 channels per stage
#define NSTG     3                  // stages per n-pair (3*256 = 768)
#define NT       (8 * NSTG)         // 24 pipeline steps per workgroup
#define LDSROW   528                // padded LDS row stride (bytes) for 256 bf16
#define LDSMAT   (32 * LDSROW)      // one 32r x 256c weight tile: 16,896 B
#define LDSBUF   (4 * LDSMAT)       // 4 matrices per stage: 67,584 B
#define SMEMSZ   (2 * LDSBUF)       // double buffer: 135,168 B

// ---------------- workspace layout (bytes) ----------------
#define XT_BYTES (B_ * PPAD * C_ * 2)          // 20,447,232
#define WT_BYTES (2 * N_ * R_ * C_ * 2)        //  6,291,456
#define Z_OFF    (2 * XT_BYTES + WT_BYTES)     // 47,185,920 (+16KB for z)

// ---------------- pass 1a: X/Y [b][c][p] f32 -> Xt [b][p][c] bf16 (zero-pad p) -----
__global__ __launch_bounds__(256) void conv_xpose(const float* __restrict__ in,
                                                  __bf16* __restrict__ out) {
  __shared__ float t[32][33];
  int b  = blockIdx.z;
  int c0 = blockIdx.y * 32;
  int p0 = blockIdx.x * 32;
  int tx = threadIdx.x, ty = threadIdx.y;
  #pragma unroll
  for (int i = 0; i < 4; ++i) {
    int c = c0 + ty + 8 * i;
    int p = p0 + tx;
    t[ty + 8 * i][tx] = (p < D2_) ? in[((size_t)b * C_ + c) * D2_ + p] : 0.f;
  }
  __syncthreads();
  #pragma unroll
  for (int i = 0; i < 4; ++i) {
    int p = p0 + ty + 8 * i;
    int c = c0 + tx;
    if (p < PPAD)
      out[((size_t)b * PPAD + p) * C_ + c] = (__bf16)t[tx][ty + 8 * i];
  }
}

// ---------------- pass 1b: w [m][c][r] f32 -> wt [m][r][c] bf16 ----------------
__global__ __launch_bounds__(256) void conv_w(const float* __restrict__ in,
                                              __bf16* __restrict__ out) {
  __shared__ float t[32][33];
  int m  = blockIdx.y;          // 0..127 (2*64 matrices)
  int c0 = blockIdx.x * 32;
  int tx = threadIdx.x, ty = threadIdx.y;
  #pragma unroll
  for (int i = 0; i < 4; ++i) {
    int c = c0 + ty + 8 * i;
    t[ty + 8 * i][tx] = in[((size_t)m * C_ + c) * R_ + tx];
  }
  __syncthreads();
  #pragma unroll
  for (int i = 0; i < 4; ++i) {
    int r = ty + 8 * i;
    out[((size_t)m * R_ + r) * C_ + c0 + tx] = (__bf16)t[tx][ty + 8 * i];
  }
}

// ---------------- fragment loaders (contiguous K per lane half, same order for A and B) ----------------
// A (16-bit 16x32): lanes 0-15 hold K{kb..kb+15}; lanes 16-31 K{kb+16..kb+31}.
__device__ inline v16bf load_a(const __bf16* p) {
  union { v16bf v; v8bf h[2]; } u;
  u.h[0] = *(const v8bf*)p;
  u.h[1] = *(const v8bf*)(p + 8);
  return u.v;
}
// B (16-bit 32x16) from LDS: lane = column; 16 contiguous K per lane (two b128).
__device__ inline v16bf load_b(const char* p) {
  union { v16bf v; v8bf h[2]; } u;
  u.h[0] = *(const v8bf*)p;
  u.h[1] = *(const v8bf*)(p + 16);
  return u.v;
}
__device__ inline v8f wmma_bf16(v16bf a, v16bf b, v8f c) {
  return __builtin_amdgcn_wmma_f32_16x16x32_bf16(false, a, false, b, (short)0, c,
                                                 false, false);
}

// async copy of one stage (4 weight tiles, 32r x 256c bf16 each) into LDS.
// 4096 x 16B segments spread over 416 lanes; ASYNCcnt tracks completion.
__device__ inline void issue_stage(const __bf16* __restrict__ Wt, int n0, int c0,
                                   unsigned ldsbase, int tid) {
  #pragma unroll 1
  for (int it = 0; it < 10; ++it) {
    int idx = it * 416 + tid;
    if (idx < 4096) {
      int mat = idx >> 10;           // 0:w0/n0 1:w0/n1 2:w1/n0 3:w1/n1
      int rem = idx & 1023;
      int r   = rem >> 5;
      int seg = rem & 31;
      const __bf16* g = Wt +
          (size_t)(((mat >> 1) * N_ + n0 + (mat & 1)) * (R_ * C_)) +
          (size_t)r * C_ + c0 + seg * 8;
      unsigned l = ldsbase + mat * LDSMAT + r * LDSROW + seg * 16;
      asm volatile("global_load_async_to_lds_b128 %0, %1, off"
                   :: "v"(l), "v"((unsigned long long)(uintptr_t)g)
                   : "memory");
    }
  }
}

// ---------------- pass 2: fused bilinear trace -> Z[b][n] ----------------
// grid (64 b, 4 n-groups), block 416 = 13 wave32 (one p-tile per wave).
// Weights double-buffered LDS via async copies; X/Y A-frags direct from L2.
__global__ __launch_bounds__(416, 1) void bilin_main(const __bf16* __restrict__ Xt,
                                                     const __bf16* __restrict__ Yt,
                                                     const __bf16* __restrict__ Wt,
                                                     float* __restrict__ Z) {
  extern __shared__ char smem[];       // SMEMSZ bytes (weight double buffer)
  __shared__ float zpart[13][16];

  int b    = blockIdx.x;
  int ng   = blockIdx.y;               // 16 n per workgroup
  int tid  = threadIdx.x;
  int wave = tid >> 5;                 // 0..12 -> p-tile
  int lane = tid & 31;
  int hi   = lane >> 4;
  int l16  = lane & 15;

  const __bf16* xrow = Xt + ((size_t)(b * PPAD + wave * 16 + l16)) * C_;
  const __bf16* yrow = Yt + ((size_t)(b * PPAD + wave * 16 + l16)) * C_;
  int aoff = hi ? 16 : 0;              // A-frag K sub-offset
  int boff = hi ? 16 : 0;              // B-frag K sub-offset (elements)

  unsigned lds0 = (unsigned)(uintptr_t)smem;   // low 32 bits = LDS byte offset

  v8f t1a0 = {}, t1a1 = {}, t1b0 = {}, t1b1 = {};
  v8f t2a0 = {}, t2a1 = {}, t2b0 = {}, t2b1 = {};

  // prologue: stage 0 in flight
  issue_stage(Wt, ng * 16 + 0, 0, lds0, tid);

  #pragma unroll 1
  for (int t = 0; t < NT; ++t) {
    int np = t / NSTG;
    int sc = t - np * NSTG;
    int n0 = ng * 16 + np * 2;

    // my stage-t copies have landed; publish across all waves
    asm volatile("s_wait_asynccnt 0x0" ::: "memory");
    __syncthreads();

    // kick off stage t+1 into the other buffer (freed by the barrier above)
    if (t + 1 < NT) {
      int np1 = (t + 1) / NSTG;
      int sc1 = (t + 1) - np1 * NSTG;
      issue_stage(Wt, ng * 16 + np1 * 2, sc1 * STAGE_C,
                  lds0 + ((t + 1) & 1) * LDSBUF, tid);
    }

    if (sc == 0) {
      t1a0 = (v8f){}; t1a1 = (v8f){}; t1b0 = (v8f){}; t1b1 = (v8f){};
      t2a0 = (v8f){}; t2a1 = (v8f){}; t2b0 = (v8f){}; t2b1 = (v8f){};
    }

    const char* bb = smem + (t & 1) * LDSBUF;
    const char* r0 = bb + (size_t)l16 * LDSROW;          // r-tile 0 column
    const char* r1 = bb + (size_t)(16 + l16) * LDSROW;   // r-tile 1 column

    #pragma unroll 2
    for (int kl = 0; kl < STAGE_CH; ++kl) {
      int cb = sc * STAGE_C + kl * 32;                   // global channel base
      v16bf ax = load_a(xrow + cb + aoff);
      v16bf ay = load_a(yrow + cb + aoff);

      int ko = (kl * 32 + boff) * 2;                     // byte offset in row
      v16bf b00 = load_b(r0 + 0 * LDSMAT + ko);          // w0, n0
      v16bf b01 = load_b(r1 + 0 * LDSMAT + ko);
      v16bf c00 = load_b(r0 + 1 * LDSMAT + ko);          // w0, n1
      v16bf c01 = load_b(r1 + 1 * LDSMAT + ko);
      v16bf b10 = load_b(r0 + 2 * LDSMAT + ko);          // w1, n0
      v16bf b11 = load_b(r1 + 2 * LDSMAT + ko);
      v16bf c10 = load_b(r0 + 3 * LDSMAT + ko);          // w1, n1
      v16bf c11 = load_b(r1 + 3 * LDSMAT + ko);

      t1a0 = wmma_bf16(ax, b00, t1a0);
      t1a1 = wmma_bf16(ax, b01, t1a1);
      t2a0 = wmma_bf16(ay, b10, t2a0);
      t2a1 = wmma_bf16(ay, b11, t2a1);
      t1b0 = wmma_bf16(ax, c00, t1b0);
      t1b1 = wmma_bf16(ax, c01, t1b1);
      t2b0 = wmma_bf16(ay, c10, t2b0);
      t2b1 = wmma_bf16(ay, c11, t2b1);
    }

    if (sc == NSTG - 1) {
      // elementwise <T1,T2> (accumulators share the C/D lane layout)
      float sa = 0.f, sb = 0.f;
      #pragma unroll
      for (int j = 0; j < 8; ++j) {
        sa += t1a0[j] * t2a0[j] + t1a1[j] * t2a1[j];
        sb += t1b0[j] * t2b0[j] + t1b1[j] * t2b1[j];
      }
      #pragma unroll
      for (int m = 16; m >= 1; m >>= 1) {
        sa += __shfl_xor(sa, m, 32);
        sb += __shfl_xor(sb, m, 32);
      }
      if (lane == 0) {
        zpart[wave][np * 2 + 0] = sa;
        zpart[wave][np * 2 + 1] = sb;
      }
    }
  }

  __syncthreads();
  if (tid < 16) {
    float s = 0.f;
    #pragma unroll
    for (int w = 0; w < 13; ++w) s += zpart[w][tid];
    Z[b * N_ + ng * 16 + tid] = s;
  }
}

// ---------------- pass 3: softmax over n (1 wave per row) ----------------
__global__ __launch_bounds__(32) void softmax_k(const float* __restrict__ Z,
                                                const float* __restrict__ bias,
                                                float* __restrict__ out) {
  int b = blockIdx.x, lane = threadIdx.x;
  const float inv = 1.f / (float)D2_;
  float z0 = Z[b * N_ + lane]      * inv + bias[lane];
  float z1 = Z[b * N_ + lane + 32] * inv + bias[lane + 32];
  float m = fmaxf(z0, z1);
  #pragma unroll
  for (int s = 16; s >= 1; s >>= 1) m = fmaxf(m, __shfl_xor(m, s, 32));
  float e0 = __expf(z0 - m), e1 = __expf(z1 - m);
  float sum = e0 + e1;
  #pragma unroll
  for (int s = 16; s >= 1; s >>= 1) sum += __shfl_xor(sum, s, 32);
  out[b * N_ + lane]      = e0 / sum;
  out[b * N_ + lane + 32] = e1 / sum;
}

extern "C" void kernel_launch(void* const* d_in, const int* in_sizes, int n_in,
                              void* d_out, int out_size, void* d_ws, size_t ws_size,
                              hipStream_t stream) {
  const float* X    = (const float*)d_in[0];  // [64,768,14,14]
  const float* Y    = (const float*)d_in[1];  // [64,768,14,14]
  const float* w    = (const float*)d_in[2];  // [2,64,768,32]
  const float* bias = (const float*)d_in[3];  // [64]
  float* out = (float*)d_out;                 // [64,64]

  char*   ws = (char*)d_ws;                   // needs ~47.2 MB
  __bf16* Xt = (__bf16*)(ws);
  __bf16* Yt = (__bf16*)(ws + XT_BYTES);
  __bf16* Wt = (__bf16*)(ws + 2 * XT_BYTES);
  float*  Zb = (float*)(ws + Z_OFF);

  dim3 tb(32, 8);
  conv_xpose<<<dim3(7, 24, B_), tb, 0, stream>>>(X, Xt);
  conv_xpose<<<dim3(7, 24, B_), tb, 0, stream>>>(Y, Yt);
  conv_w<<<dim3(24, 2 * N_), tb, 0, stream>>>(w, Wt);
  bilin_main<<<dim3(B_, 4), 416, SMEMSZ, stream>>>(Xt, Yt, Wt, Zb);
  softmax_k<<<B_, 32, 0, stream>>>(Zb, bias, out);
}
